// GCN_22668837388503
// MI455X (gfx1250) — matrix operand order repaired and checked by reading, not compiled, not run on hardware
//
#include <hip/hip_runtime.h>

typedef _Float16 half_t;
typedef __attribute__((ext_vector_type(16))) _Float16 v16h;
typedef __attribute__((ext_vector_type(8)))  float    v8f;

#define BN_EPS 1e-5f

__device__ __forceinline__ v8f wmma_f16(v16h a, v16h b, v8f c) {
  // D = A(16x32 f16) x B(32x16 f16) + C(16x16 f32)
  return __builtin_amdgcn_wmma_f32_16x16x32_f16(false, a, false, b, (short)0, c,
                                                false, false);
}

// ---------------------------------------------------------------- copy z = h
__global__ void copy_kernel(float4* __restrict__ dst, const float4* __restrict__ src,
                            long long n4) {
  long long i = (long long)blockIdx.x * blockDim.x + threadIdx.x;
  if (i < n4) dst[i] = src[i];
}

// ------------------------------------------------------- edge scatter-add agg
// z[dst] += h[src]; thread handles one float4 chunk of one edge.
template <int F>
__global__ void agg_kernel(const float* __restrict__ h, const int* __restrict__ edge,
                           float* __restrict__ z, long long E) {
  const int PARTS = F / 4;
  long long i = (long long)blockIdx.x * blockDim.x + threadIdx.x;
  long long total = E * (long long)PARTS;
  if (i >= total) return;
  long long e = i / PARTS;
  int p = (int)(i - e * (long long)PARTS);
  int s = edge[e];      // src
  int d = edge[E + e];  // dst
  float4 v = ((const float4*)(h + (size_t)s * F))[p];
  float* zp = z + (size_t)d * F + (size_t)p * 4;
  unsafeAtomicAdd(zp + 0, v.x);
  unsafeAtomicAdd(zp + 1, v.y);
  unsafeAtomicAdd(zp + 2, v.z);
  unsafeAtomicAdd(zp + 3, v.w);
}

// --------------------------------------------- per-layer weight / BN prep
// Pre-swizzles weights into WMMA B-fragment layout (f16) and folds BN+bias
// into per-channel scale/shift:  y = relu(lin(z))  ->  relu(z*W * s + t)
// B frag layout (32x16 f16): lane l holds col = l&15, halves e=0..15 are
// K = (l<16 ? 0 : 16) + e   (contiguous k order chosen here; A frags use the same)
template <int DIN, int DOUT>
__global__ void prep_layer(const float* __restrict__ wa, const float* __restrict__ ba,
                           const float* __restrict__ g,  const float* __restrict__ be,
                           const float* __restrict__ rm, const float* __restrict__ rv,
                           const float* __restrict__ wb,
                           half_t* __restrict__ waF, half_t* __restrict__ wbF,
                           float* __restrict__ scale, float* __restrict__ shift) {
  int tid = threadIdx.x;
  if (tid < 32) {
    float s = g[tid] * rsqrtf(rv[tid] + BN_EPS);
    scale[tid] = s;
    shift[tid] = (ba[tid] - rm[tid]) * s + be[tid];
  }
  const int FR_A = (DIN / 32) * 2;  // chunks * col-halves
  for (int idx = tid; idx < FR_A * 512; idx += blockDim.x) {
    int frag = idx >> 9, w = idx & 511, lane = w >> 4, e = w & 15;
    int chunk = frag >> 1, jh = frag & 1;
    int col = jh * 16 + (lane & 15);
    int k = chunk * 32 + ((lane < 16) ? 0 : 16) + e;
    waF[idx] = (half_t)wa[k * 32 + col];
  }
  const int FR_B = DOUT / 16;
  for (int idx = tid; idx < FR_B * 512; idx += blockDim.x) {
    int frag = idx >> 9, w = idx & 511, lane = w >> 4, e = w & 15;
    int col = frag * 16 + (lane & 15);
    int k = ((lane < 16) ? 0 : 16) + e;
    wbF[idx] = (half_t)wb[k * DOUT + col];
  }
}

// -------------------------------------------------- fused MLP (WMMA, 1 wave)
// out = relu( relu(BN(z @ wa + ba)) @ wb + bb ), 16 nodes per wave.
// A frag layout (16x32 f16): lane l holds row M = l&15; halves 0..15 are
// K = base+0..15, base = (l<16 ? 0 : 16).
template <int DIN, int DOUT>
__global__ void __launch_bounds__(32)
mlp_kernel(const float* __restrict__ z, const half_t* __restrict__ waF,
           const float* __restrict__ scale, const float* __restrict__ shift,
           const half_t* __restrict__ wbF, const float* __restrict__ bb,
           float* __restrict__ out, int nTiles) {
  __shared__ half_t mid[16][32];
  int t = blockIdx.x;
  if (t >= nTiles) return;
  const int l = threadIdx.x & 31;
  const int m = l & 15;
  const bool hi = l >= 16;
  const int rowBase = t * 16;

  v8f acc0 = {}, acc1 = {};
  constexpr int CHUNKS = DIN / 32;
#pragma unroll
  for (int c = 0; c < CHUNKS; ++c) {
    const float* zr = z + (size_t)(rowBase + m) * DIN + c * 32 + (hi ? 16 : 0);
    v16h a;
#pragma unroll
    for (int e = 0; e < 8; ++e) a[e] = (half_t)zr[e];
#pragma unroll
    for (int e = 0; e < 8; ++e) a[8 + e] = (half_t)zr[8 + e];
    v16h b0 = *(const v16h*)(waF + (size_t)(c * 2 + 0) * 512 + l * 16);
    v16h b1 = *(const v16h*)(waF + (size_t)(c * 2 + 1) * 512 + l * 16);
    acc0 = wmma_f16(a, b0, acc0);
    acc1 = wmma_f16(a, b1, acc1);
  }

  // BN fold + inner ReLU; C layout: VGPR r -> row rb+r, col m (acc0) / 16+m (acc1)
  const float s0 = scale[m], t0 = shift[m];
  const float s1 = scale[16 + m], t1 = shift[16 + m];
  const int rb = hi ? 8 : 0;
#pragma unroll
  for (int r = 0; r < 8; ++r) {
    float v0 = acc0[r] * s0 + t0;
    float v1 = acc1[r] * s1 + t1;
    mid[rb + r][m]      = (half_t)(v0 > 0.f ? v0 : 0.f);
    mid[rb + r][16 + m] = (half_t)(v1 > 0.f ? v1 : 0.f);
  }
  __syncthreads();  // single-wave WG: barrier is NOP, waits DScnt

  // second GEMM: A from LDS transpose
  v16h a2;
  const int k2 = hi ? 16 : 0;
#pragma unroll
  for (int e = 0; e < 8; ++e) a2[e] = mid[m][k2 + e];
#pragma unroll
  for (int e = 0; e < 8; ++e) a2[8 + e] = mid[m][k2 + 8 + e];

  v8f d0 = {};
  v16h w0 = *(const v16h*)(wbF + (size_t)l * 16);
  d0 = wmma_f16(a2, w0, d0);
  const float bb0 = bb[m];
  if constexpr (DOUT == 32) {
    v8f d1 = {};
    v16h w1 = *(const v16h*)(wbF + 512 + (size_t)l * 16);
    d1 = wmma_f16(a2, w1, d1);
    const float bb1 = bb[16 + m];
#pragma unroll
    for (int r = 0; r < 8; ++r) {
      int row = rowBase + rb + r;
      float v0 = d0[r] + bb0;
      float v1 = d1[r] + bb1;
      out[(size_t)row * 32 + m]      = v0 > 0.f ? v0 : 0.f;
      out[(size_t)row * 32 + 16 + m] = v1 > 0.f ? v1 : 0.f;
    }
  } else {  // DOUT == 16
#pragma unroll
    for (int r = 0; r < 8; ++r) {
      int row = rowBase + rb + r;
      float v0 = d0[r] + bb0;
      out[(size_t)row * 16 + m] = v0 > 0.f ? v0 : 0.f;
    }
  }
}

// ------------------------------------------------------------- pool + softmax
__global__ void zero_kernel(float* __restrict__ p, int n) {
  int i = blockIdx.x * blockDim.x + threadIdx.x;
  if (i < n) p[i] = 0.f;
}

__global__ void pool_kernel(const float* __restrict__ h, const int* __restrict__ batch,
                            float* __restrict__ pooled, int N) {
  int i = blockIdx.x * blockDim.x + threadIdx.x;
  if (i >= N * 16) return;
  int n = i >> 4, c = i & 15;
  unsafeAtomicAdd(&pooled[batch[n] * 16 + c], h[i]);
}

__global__ void softmax_kernel(const float* __restrict__ pooled, float* __restrict__ out) {
  int g = blockIdx.x * blockDim.x + threadIdx.x;
  if (g >= 256) return;
  float v[16], mx = -3.4e38f;
#pragma unroll
  for (int c = 0; c < 16; ++c) { v[c] = pooled[g * 16 + c]; mx = v[c] > mx ? v[c] : mx; }
  float s = 0.f;
#pragma unroll
  for (int c = 0; c < 16; ++c) { v[c] = __expf(v[c] - mx); s += v[c]; }
  float inv = 1.f / s;
#pragma unroll
  for (int c = 0; c < 16; ++c) out[g * 16 + c] = v[c] * inv;
}

// ---------------------------------------------------------------------------
extern "C" void kernel_launch(void* const* d_in, const int* in_sizes, int n_in,
                              void* d_out, int out_size, void* d_ws, size_t ws_size,
                              hipStream_t stream) {
  const float* x    = (const float*)d_in[0];
  const int* edge   = (const int*)d_in[1];
  const int* batch  = (const int*)d_in[2];
  const long long E = in_sizes[1] / 2;
  const int N       = in_sizes[0] / 128;
  const int nTiles  = N / 16;

  auto W = [&](int layer, int slot) -> const float* {
    // slots: 0 wa, 1 ba, 2 g, 3 be, 4 rm, 5 rv, 6 wb, 7 bb
    return (const float*)d_in[3 + (layer - 1) * 8 + slot];
  };

  char* ws = (char*)d_ws;
  half_t* wfr    = (half_t*)(ws);                 // 25 frags * 1KB
  float*  ss     = (float*)(ws + (64 << 10));     // 5 * (32 scale + 32 shift)
  float*  pooled = (float*)(ws + (80 << 10));     // 256*16
  float*  zbuf   = (float*)(ws + (128 << 10));    // N*128 f32
  size_t  zB     = (size_t)N * 128 * sizeof(float);
  float*  hA     = (float*)((char*)zbuf + zB);    // N*32 f32
  size_t  hB_    = (size_t)N * 32 * sizeof(float);
  float*  hB     = (float*)((char*)hA + hB_);     // N*32 f32

  half_t* waF[5] = { wfr + 0 * 512, wfr + 10 * 512, wfr + 14 * 512,
                     wfr + 18 * 512, wfr + 22 * 512 };
  half_t* wbF[5] = { wfr + 8 * 512, wfr + 12 * 512, wfr + 16 * 512,
                     wfr + 20 * 512, wfr + 24 * 512 };

  // ---- prep (weight swizzle to f16 fragments, BN fold) ----
  prep_layer<128, 32><<<1, 256, 0, stream>>>(W(1,0), W(1,1), W(1,2), W(1,3), W(1,4),
                                             W(1,5), W(1,6), waF[0], wbF[0],
                                             ss + 0, ss + 32);
  prep_layer<32, 32><<<1, 256, 0, stream>>>(W(2,0), W(2,1), W(2,2), W(2,3), W(2,4),
                                            W(2,5), W(2,6), waF[1], wbF[1],
                                            ss + 64, ss + 96);
  prep_layer<32, 32><<<1, 256, 0, stream>>>(W(3,0), W(3,1), W(3,2), W(3,3), W(3,4),
                                            W(3,5), W(3,6), waF[2], wbF[2],
                                            ss + 128, ss + 160);
  prep_layer<32, 32><<<1, 256, 0, stream>>>(W(4,0), W(4,1), W(4,2), W(4,3), W(4,4),
                                            W(4,5), W(4,6), waF[3], wbF[3],
                                            ss + 192, ss + 224);
  prep_layer<32, 16><<<1, 256, 0, stream>>>(W(5,0), W(5,1), W(5,2), W(5,3), W(5,4),
                                            W(5,5), W(5,6), waF[4], wbF[4],
                                            ss + 256, ss + 288);

  // ---- layer 1 (DIN=128) ----
  {
    long long n4 = (long long)N * 128 / 4;
    copy_kernel<<<(int)((n4 + 255) / 256), 256, 0, stream>>>((float4*)zbuf,
                                                             (const float4*)x, n4);
    long long tot = E * 32;
    agg_kernel<128><<<(int)((tot + 255) / 256), 256, 0, stream>>>(x, edge, zbuf, E);
    mlp_kernel<128, 32><<<nTiles, 32, 0, stream>>>(zbuf, waF[0], ss + 0, ss + 32,
                                                   wbF[0], W(1, 7), hA, nTiles);
  }

  // ---- layers 2..4 (DIN=DOUT=32), ping-pong hA/hB ----
  const float* hin = hA;
  float* hout = hB;
  for (int L = 2; L <= 4; ++L) {
    long long n4 = (long long)N * 32 / 4;
    copy_kernel<<<(int)((n4 + 255) / 256), 256, 0, stream>>>((float4*)zbuf,
                                                             (const float4*)hin, n4);
    long long tot = E * 8;
    agg_kernel<32><<<(int)((tot + 255) / 256), 256, 0, stream>>>(hin, edge, zbuf, E);
    int li = L - 1;
    mlp_kernel<32, 32><<<nTiles, 32, 0, stream>>>(zbuf, waF[li], ss + li * 64,
                                                  ss + li * 64 + 32, wbF[li],
                                                  W(L, 7), hout, nTiles);
    float* tmp = (float*)hin; hin = hout; hout = tmp;
  }

  // ---- layer 5 (DOUT=16) ----
  {
    long long n4 = (long long)N * 32 / 4;
    copy_kernel<<<(int)((n4 + 255) / 256), 256, 0, stream>>>((float4*)zbuf,
                                                             (const float4*)hin, n4);
    long long tot = E * 8;
    agg_kernel<32><<<(int)((tot + 255) / 256), 256, 0, stream>>>(hin, edge, zbuf, E);
    mlp_kernel<32, 16><<<nTiles, 32, 0, stream>>>(zbuf, waF[4], ss + 256, ss + 288,
                                                  wbF[4], W(5, 7), hout, nTiles);
  }
  const float* hfinal = hout;  // N x 16

  // ---- pool + softmax ----
  zero_kernel<<<16, 256, 0, stream>>>(pooled, 256 * 16);
  pool_kernel<<<(N * 16 + 255) / 256, 256, 0, stream>>>(hfinal, batch, pooled, N);
  softmax_kernel<<<1, 256, 0, stream>>>(pooled, (float*)d_out);
}
